// MegalodonAttention_12455405158586
// MI455X (gfx1250) — matrix operand order repaired and checked by reading, not compiled, not run on hardware
//
#include <hip/hip_runtime.h>
#include <hip/hip_bf16.h>
#include <cstdint>
#include <cstddef>

// ---------------------------------------------------------------------------
// CDNA5 WMMA (wave32): D(16x16,f32) = A(16x32,bf16) x B(32x16,bf16) + C
// ---------------------------------------------------------------------------
typedef __attribute__((ext_vector_type(16))) __bf16 v16bf;
typedef __attribute__((ext_vector_type(8)))  __bf16 bf16x8;
typedef __attribute__((ext_vector_type(8)))  float  v8f;

#define L_   4096
#define B_   8
#define D_   1024
#define Z_   256
#define H_   2048
#define N_   64
#define C_   512
#define ROWS (L_*B_)          // 32768, activation row index r = t*B + b
#define NCH  (L_/C_)          // 8 chunks
#define MXW  4352             // D+Z+H+D

__device__ __forceinline__ float sigmoidf_(float x){ return 1.0f/(1.0f + __expf(-x)); }
__device__ __forceinline__ float siluf_(float x){ return x * sigmoidf_(x); }
__device__ __forceinline__ v8f zero8(){ v8f z; for (int i=0;i<8;i++) z[i]=0.0f; return z; }

// CDNA5 async global->LDS copy (ASYNCcnt path); 16B per lane.
__device__ __forceinline__ void async_ld_b128(uint32_t lds_off, const __bf16* g){
  asm volatile("global_load_async_to_lds_b128 %0, %1, off"
               :: "v"(lds_off), "v"((unsigned long long)(uintptr_t)g)
               : "memory");
}
__device__ __forceinline__ void wait_async0(){
  asm volatile("s_wait_asynccnt 0" ::: "memory");
}

// A fragment (16x32) from LDS tile (row-major, stride LDA_ elems).
// lanes0-15: m=lane, elems 0..7 -> K 0..7, 8..15 -> K 16..23;
// lanes16-31: elems 0..7 -> K 8..15, 8..15 -> K 24..31.  (two b128 reads)
#define LDA_ 40
__device__ __forceinline__ v16bf lds_afrag(const __bf16* sA, int row0, int lane){
  const __bf16* p = sA + (size_t)(row0 + (lane & 15)) * LDA_ + ((lane >> 4) * 8);
  bf16x8 lo = *(const bf16x8*)p;
  bf16x8 hi = *(const bf16x8*)(p + 16);
  v16bf a;
#pragma unroll
  for (int e = 0; e < 8; ++e) { a[e] = lo[e]; a[e+8] = hi[e]; }
  return a;
}
// B fragment (32x16) straight from K-contiguous global memory.
// base points at column n, element K index kb (= k0 + (lane>>4)*8).
__device__ __forceinline__ v16bf gbl_bfrag(const __bf16* base){
  bf16x8 lo = *(const bf16x8*)base;
  bf16x8 hi = *(const bf16x8*)(base + 16);
  v16bf b;
#pragma unroll
  for (int e = 0; e < 8; ++e) { b[e] = lo[e]; b[e+8] = hi[e]; }
  return b;
}

// ---------------------------------------------------------------------------
// 128x128 block, 8-wave (2x4) GEMM mainloop.  A:[M,K] bf16 row-major (LDS via
// async copy).  WT:[N,K] bf16 row-major (= W transposed; B-frags direct from
// global, L2-resident).  Wave owns 64x32 = 4(m) x 2(n) accumulators.
// ---------------------------------------------------------------------------
#define BM 128
#define BN 128
#define BK 32

struct Acc8 { v8f c[4][2]; };

__device__ __forceinline__ void gemm_tiles(
    const __bf16* __restrict__ A, const __bf16* __restrict__ WT,
    int K, int m0, int n0, __bf16* sA, Acc8& acc)
{
  const int tid  = threadIdx.x;
  const int wave = tid >> 5, lane = tid & 31;
  const int wrow = (wave >> 2) * 64;
  const int wcol = (wave & 3) * 32;
#pragma unroll
  for (int i = 0; i < 4; ++i)
#pragma unroll
    for (int j = 0; j < 2; ++j) acc.c[i][j] = zero8();

  // A tile staging map: 2 threads per row, 16 elems each
  const int ar = tid >> 1, ac = (tid & 1) * 16;
  const uint32_t ldsA0 = (uint32_t)(uintptr_t)(sA + (size_t)ar * LDA_ + ac);
  const __bf16* gA = A + (size_t)(m0 + ar) * K + ac;

  // B fragment base pointers (column n, K-major)
  const __bf16* bp[2];
#pragma unroll
  for (int j = 0; j < 2; ++j)
    bp[j] = WT + (size_t)(n0 + wcol + j*16 + (lane & 15)) * K + ((lane >> 4) * 8);

  for (int k0 = 0; k0 < K; k0 += BK) {
    async_ld_b128(ldsA0,      gA + k0);
    async_ld_b128(ldsA0 + 16, gA + k0 + 8);
    if (k0 + BK < K)
      __builtin_prefetch((const void*)(gA + k0 + BK), 0, 1);
    wait_async0();
    __syncthreads();
    v16bf af[4], bfr[2];
#pragma unroll
    for (int i = 0; i < 4; ++i) af[i]  = lds_afrag(sA, wrow + i*16, lane);
#pragma unroll
    for (int j = 0; j < 2; ++j) bfr[j] = gbl_bfrag(bp[j] + k0);
#pragma unroll
    for (int i = 0; i < 4; ++i)
#pragma unroll
      for (int j = 0; j < 2; ++j)
        acc.c[i][j] = __builtin_amdgcn_wmma_f32_16x16x32_bf16(
            false, af[i], false, bfr[j], (short)0, acc.c[i][j], false, false);
    __syncthreads();
  }
}

// C/D layout: VGPR r -> row = r + 8*(lane>=16); col = lane&15.
// Variadic so commas in the body survive preprocessing.
#define EPILOGUE_LOOP(...)                                                     \
  { const int lane = threadIdx.x & 31, wave = threadIdx.x >> 5;                \
    const int wrow = (wave >> 2) * 64, wcol = (wave & 3) * 32;                 \
    _Pragma("unroll") for (int i = 0; i < 4; ++i)                              \
    _Pragma("unroll") for (int j = 0; j < 2; ++j)                              \
    _Pragma("unroll") for (int r = 0; r < 8; ++r) {                            \
      const int row = m0 + wrow + i*16 + ((lane >> 4) * 8) + r;                \
      const int col = n0 + wcol + j*16 + (lane & 15);                          \
      __VA_ARGS__ } }

// ---------------------------------------------------------------------------
// Elementwise / prep kernels
// ---------------------------------------------------------------------------
__global__ void f2bf_t_kernel(const float* __restrict__ in, __bf16* __restrict__ out,
                              int Kdim, int Ndim){
  const size_t i = (size_t)blockIdx.x * 256 + threadIdx.x;
  if (i >= (size_t)Kdim * Ndim) return;
  const int k = (int)(i / Ndim);
  const int n = (int)(i % Ndim);
  out[(size_t)n * Kdim + k] = (__bf16)in[i];      // W[k][n] -> WT[n][k]
}

__global__ void scalenorm_kernel(const float* __restrict__ x, const float* __restrict__ g,
                                 float* __restrict__ xnF, __bf16* __restrict__ xnB){
  const int r = blockIdx.x, tid = threadIdx.x;
  __shared__ float red[256];
  const float* xr = x + (size_t)r * D_;
  float s = 0.f;
  for (int c = tid; c < D_; c += 256) { float v = xr[c]; s += v * v; }
  red[tid] = s; __syncthreads();
  for (int off = 128; off > 0; off >>= 1){ if (tid < off) red[tid] += red[tid+off]; __syncthreads(); }
  const float scale = rsqrtf(red[0] * (1.0f / D_) + 1e-6f) * g[0];
  for (int c = tid; c < D_; c += 256) {
    float v = xr[c] * scale;
    xnF[(size_t)r * D_ + c] = v;
    xnB[(size_t)r * D_ + c] = (__bf16)v;
  }
}

// value = silu(xn @ Wv + bv), stored K-major for attention PV: vT[b][h][t]
__global__ void value_gemm_kernel(const __bf16* __restrict__ A, const __bf16* __restrict__ WvT,
                                  const float* __restrict__ bv, __bf16* __restrict__ vT){
  __shared__ __bf16 sA[BM * LDA_];
  Acc8 acc;
  const int m0 = blockIdx.x * BM, n0 = blockIdx.y * BN;
  gemm_tiles(A, WvT, D_, m0, n0, sA, acc);
  EPILOGUE_LOOP(
    const float v = acc.c[i][j][r] + bv[col];
    const int t = row >> 3;                       // row = t*B + b
    const int bb = row & 7;
    vT[((size_t)bb * H_ + col) * L_ + t] = (__bf16)siluf_(v);
  )
}

// EMA via 64 parallel first-order recurrences; fused mx_in = silu(ema + xn*res_w).
__global__ void ema_scan_kernel(const float* __restrict__ xnF,
                                const float* __restrict__ damping, const float* __restrict__ decay,
                                const float* __restrict__ expansion, const float* __restrict__ kproj,
                                const float* __restrict__ res_w, __bf16* __restrict__ mx_inB){
  const int wave = threadIdx.x >> 5, lane = threadIdx.x & 31;
  const int pair = blockIdx.x * 8 + wave;     // B_*D_ = 8192 pairs
  const int b = pair >> 10, d = pair & 1023;
  const float inv_sqrtN = 0.125f;
  float c0, w0, c1, w1;
  {
    int n = lane;
    float p = sigmoidf_(damping[d*N_ + n]);
    w0 = 1.0f - p * sigmoidf_(decay[d*N_ + n]);
    c0 = p * expansion[d*N_ + n] * kproj[d*N_ + n] * inv_sqrtN;
    n = lane + 32;
    p = sigmoidf_(damping[d*N_ + n]);
    w1 = 1.0f - p * sigmoidf_(decay[d*N_ + n]);
    c1 = p * expansion[d*N_ + n] * kproj[d*N_ + n] * inv_sqrtN;
  }
  const float rw = res_w[d];
  float s0 = 0.f, s1 = 0.f;
  for (int t = 0; t < L_; ++t) {
    const size_t idx = ((size_t)t * B_ + b) * D_ + d;
    const float xv = xnF[idx];
    s0 = w0 * s0 + xv;
    s1 = w1 * s1 + xv;
    float y = c0 * s0 + c1 * s1;
#pragma unroll
    for (int off = 16; off > 0; off >>= 1) y += __shfl_xor(y, off, 32);
    if (lane == 0) mx_inB[idx] = (__bf16)siluf_(y + xv * rw);
  }
}

__global__ void mx_gemm_kernel(const __bf16* __restrict__ A, const __bf16* __restrict__ WmxT,
                               const float* __restrict__ bmx,
                               __bf16* __restrict__ rwB, __bf16* __restrict__ qkB,
                               __bf16* __restrict__ gateB, __bf16* __restrict__ interB){
  __shared__ __bf16 sA[BM * LDA_];
  Acc8 acc;
  const int m0 = blockIdx.x * BM, n0 = blockIdx.y * BN;
  gemm_tiles(A, WmxT, D_, m0, n0, sA, acc);
  EPILOGUE_LOOP(
    const float v = acc.c[i][j][r] + bmx[col];
    if      (col < 1024) rwB  [(size_t)row*1024 + col        ] = (__bf16)sigmoidf_(v);
    else if (col < 1280) qkB  [(size_t)row*256  + (col-1024) ] = (__bf16)siluf_(v);
    else if (col < 3328) gateB[(size_t)row*2048 + (col-1280) ] = (__bf16)siluf_(v);
    else                 interB[(size_t)row*1024 + (col-3328)] = (__bf16)v;
  )
}

__global__ void qk_prep_kernel(const __bf16* __restrict__ qkB,
                               const float* __restrict__ qk_w, const float* __restrict__ qk_b,
                               __bf16* __restrict__ qB, __bf16* __restrict__ kB){
  const size_t i = (size_t)blockIdx.x * 256 + threadIdx.x;   // ROWS*Z_ elements
  const int z = (int)(i & (Z_ - 1));
  const float v = (float)qkB[i];
  qB[i] = (__bf16)(v * qk_w[z]      + qk_b[z]);
  kB[i] = (__bf16)(v * qk_w[Z_ + z] + qk_b[Z_ + z]);
}

// rotary bias: bias[i,j] = sum_z ra[i,z]*rb[j,z]; alpha/beta broadcast over rows.
__global__ void bias_kernel(const float* __restrict__ alpha, const float* __restrict__ beta,
                            float* __restrict__ biasF){
  const int idx = blockIdx.x * 256 + threadIdx.x;            // C_*C_ threads
  const int i = idx >> 9, j = idx & 511;
  const float kfac = -9.210340371976184f / 128.0f;           // -ln(10000)/half
  float acc = 0.f;
  for (int z = 0; z < 128; ++z) {
    const float f = __expf((float)z * kfac);
    float si, ci, sj, cj;
    __sincosf((float)i * f, &si, &ci);
    __sincosf((float)j * f, &sj, &cj);
    const float a1 = alpha[z], a2 = alpha[128 + z];
    const float b1 = beta[z],  b2 = beta[128 + z];
    acc += (a1*ci - a2*si) * (b1*cj - b2*sj) + (a2*ci + a1*si) * (b2*cj + b1*sj);
  }
  biasF[idx] = acc;
}

// ---------------------------------------------------------------------------
// Fused per-chunk attention.  grid = (64 chunks, 4 query-tiles of 64 rows).
// Phase A: S = QK^T/16 + bias + causal mask, full-row softmax, P -> LDS bf16.
// Phase B: O = P @ V with V fragments streamed K-contiguous from vT[b][h][t]
// (2 x b128 per fragment, L2-resident); fused *rowinv*gate -> wg bf16.
// ---------------------------------------------------------------------------
#define PLD 520     // P row stride (bf16)
#define ATTN_LDS_BYTES (64*PLD*2 + 512*4 + 512*4 + 64*4 + 64*4)

__global__ void attn_kernel(const __bf16* __restrict__ qB, const __bf16* __restrict__ kB,
                            const __bf16* __restrict__ vT, const __bf16* __restrict__ gateB,
                            const float* __restrict__ biasF, __bf16* __restrict__ wgB){
  extern __shared__ char smem[];
  __bf16* sP     = (__bf16*)smem;                       // 64 x PLD
  float*  pmax   = (float*)(smem + 64*PLD*2);           // [8][64]
  float*  psum   = pmax + 512;                          // [8][64]
  float*  rowred = psum + 512;                          // [64]
  float*  rowinv = rowred + 64;                         // [64]

  const int tid = threadIdx.x, wave = tid >> 5, lane = tid & 31;
  const int lrow = lane & 15, lhi = lane >> 4;
  const int bc = blockIdx.x, b = bc >> 3, ch = bc & 7;
  const int qrow0 = blockIdx.y * 64;

  // ---------------- Phase A: scores + softmax ----------------
  v8f sc[4][4];
#pragma unroll
  for (int i = 0; i < 4; ++i)
#pragma unroll
    for (int j = 0; j < 4; ++j) sc[i][j] = zero8();
  const int col0 = wave * 64;

  for (int k0 = 0; k0 < Z_; k0 += 32) {
    v16bf af[4], bfr[4];
#pragma unroll
    for (int i = 0; i < 4; ++i) {                 // Q fragments (A layout)
      const int row = qrow0 + i*16 + lrow;
      const size_t r = ((size_t)(ch*C_ + row)) * B_ + b;
      const __bf16* p = qB + r * Z_ + k0 + lhi*8;
      bf16x8 lo = *(const bf16x8*)p;
      bf16x8 hi = *(const bf16x8*)(p + 16);
#pragma unroll
      for (int e = 0; e < 8; ++e) { af[i][e] = lo[e]; af[i][e+8] = hi[e]; }
    }
#pragma unroll
    for (int j = 0; j < 4; ++j) {                 // K^T fragments (B layout)
      const int key = col0 + j*16 + lrow;
      const size_t r = ((size_t)(ch*C_ + key)) * B_ + b;
      bfr[j] = gbl_bfrag(kB + r * Z_ + k0 + lhi*8);
    }
#pragma unroll
    for (int i = 0; i < 4; ++i)
#pragma unroll
      for (int j = 0; j < 4; ++j)
        sc[i][j] = __builtin_amdgcn_wmma_f32_16x16x32_bf16(
            false, af[i], false, bfr[j], (short)0, sc[i][j], false, false);
  }

  // scale + rotary bias + causal mask + partial row max
#pragma unroll
  for (int i = 0; i < 4; ++i)
#pragma unroll
    for (int r = 0; r < 8; ++r) {
      const int rl  = i*16 + lhi*8 + r;           // qtile-local row
      const int row = qrow0 + rl;                 // chunk-local row
      float mx = -1e30f;
#pragma unroll
      for (int j = 0; j < 4; ++j) {
        const int colc = col0 + j*16 + lrow;
        float s = sc[i][j][r] * 0.0625f + biasF[row * C_ + colc];
        if (colc > row) s = -1e9f;
        sc[i][j][r] = s;
        mx = fmaxf(mx, s);
      }
      for (int m = 8; m >= 1; m >>= 1) mx = fmaxf(mx, __shfl_xor(mx, m, 32));
      if (lrow == 0) pmax[wave*64 + rl] = mx;
    }
  __syncthreads();
  if (tid < 64) {
    float m = -1e30f;
    for (int w = 0; w < 8; ++w) m = fmaxf(m, pmax[w*64 + tid]);
    rowred[tid] = m;
  }
  __syncthreads();
#pragma unroll
  for (int i = 0; i < 4; ++i)
#pragma unroll
    for (int r = 0; r < 8; ++r) {
      const int rl = i*16 + lhi*8 + r;
      const float rm = rowred[rl];
      float ps = 0.f;
#pragma unroll
      for (int j = 0; j < 4; ++j) {
        const float p = __expf(sc[i][j][r] - rm);
        ps += p;
        sP[(size_t)rl * PLD + col0 + j*16 + lrow] = (__bf16)p;
      }
      for (int m = 8; m >= 1; m >>= 1) ps += __shfl_xor(ps, m, 32);
      if (lrow == 0) psum[wave*64 + rl] = ps;
    }
  __syncthreads();
  if (tid < 64) {
    float s = 0.f;
    for (int w = 0; w < 8; ++w) s += psum[w*64 + tid];
    rowinv[tid] = 1.0f / s;
  }
  __syncthreads();

  // ---------------- Phase B: O = P @ V ----------------
  for (int nc = 0; nc < 4; ++nc) {
    const int wcol = wave * 256 + nc * 64;        // wave's 64-col slab in H
    v8f oc[4][4];
#pragma unroll
    for (int i = 0; i < 4; ++i)
#pragma unroll
      for (int j = 0; j < 4; ++j) oc[i][j] = zero8();

    // V fragment base pointers: column h, keys contiguous
    const __bf16* vp[4];
#pragma unroll
    for (int j = 0; j < 4; ++j) {
      const int hcol = wcol + j*16 + lrow;
      vp[j] = vT + ((size_t)b * H_ + hcol) * L_ + ch*C_ + lhi*8;
    }
    for (int ks = 0; ks < 16; ++ks) {
      v16bf af[4], bfr[4];
#pragma unroll
      for (int i = 0; i < 4; ++i) {               // P fragments (A layout)
        const __bf16* p = sP + (size_t)(i*16 + lrow) * PLD + ks*32 + lhi*8;
        bf16x8 lo = *(const bf16x8*)p;
        bf16x8 hi = *(const bf16x8*)(p + 16);
#pragma unroll
        for (int e = 0; e < 8; ++e) { af[i][e] = lo[e]; af[i][e+8] = hi[e]; }
      }
#pragma unroll
      for (int j = 0; j < 4; ++j) bfr[j] = gbl_bfrag(vp[j] + ks*32);
#pragma unroll
      for (int i = 0; i < 4; ++i)
#pragma unroll
        for (int j = 0; j < 4; ++j)
          oc[i][j] = __builtin_amdgcn_wmma_f32_16x16x32_bf16(
              false, af[i], false, bfr[j], (short)0, oc[i][j], false, false);
    }
    // epilogue: normalize + fuse gate, store wg = weighted*gate (bf16)
#pragma unroll
    for (int i = 0; i < 4; ++i)
#pragma unroll
      for (int j = 0; j < 4; ++j)
#pragma unroll
        for (int r = 0; r < 8; ++r) {
          const int rl  = i*16 + lhi*8 + r;
          const int row = qrow0 + rl;
          const int hcol = wcol + j*16 + lrow;
          const size_t gr = ((size_t)(ch*C_ + row)) * B_ + b;
          const float o  = oc[i][j][r] * rowinv[rl];
          const float gt = (float)gateB[gr * H_ + hcol];
          wgB[gr * H_ + hcol] = (__bf16)(o * gt);
        }
  }
}

__global__ void frame_gemm_kernel(const __bf16* __restrict__ A, const __bf16* __restrict__ WhT,
                                  const float* __restrict__ bh, const __bf16* __restrict__ interB,
                                  const __bf16* __restrict__ rwB, const float* __restrict__ xin,
                                  float* __restrict__ outp){
  __shared__ __bf16 sA[BM * LDA_];
  Acc8 acc;
  const int m0 = blockIdx.x * BM, n0 = blockIdx.y * BN;
  gemm_tiles(A, WhT, H_, m0, n0, sA, acc);
  EPILOGUE_LOOP(
    float f = acc.c[i][j][r] + bh[col] + (float)interB[(size_t)row*D_ + col];
    f = siluf_(f);
    const float xv = xin[(size_t)row*D_ + col];
    const float rw = (float)rwB[(size_t)row*D_ + col];
    outp[(size_t)row*D_ + col] = xv + rw * (f - xv);
  )
}

// ---------------------------------------------------------------------------
// Host launcher
// ---------------------------------------------------------------------------
static inline size_t align256(size_t x){ return (x + 255) & ~(size_t)255; }

extern "C" void kernel_launch(void* const* d_in, const int* in_sizes, int n_in,
                              void* d_out, int out_size, void* d_ws, size_t ws_size,
                              hipStream_t stream) {
  const float* x        = (const float*)d_in[0];
  const float* g        = (const float*)d_in[1];
  const float* Wv       = (const float*)d_in[2];
  const float* bv       = (const float*)d_in[3];
  const float* Wmx      = (const float*)d_in[4];
  const float* bmx      = (const float*)d_in[5];
  const float* Wh       = (const float*)d_in[6];
  const float* bh       = (const float*)d_in[7];
  const float* qk_w     = (const float*)d_in[8];
  const float* qk_b     = (const float*)d_in[9];
  const float* damping  = (const float*)d_in[10];
  const float* decay    = (const float*)d_in[11];
  const float* expans   = (const float*)d_in[12];
  const float* kproj    = (const float*)d_in[13];
  const float* res_w    = (const float*)d_in[14];
  const float* alpha    = (const float*)d_in[15];
  const float* beta     = (const float*)d_in[16];
  float* out = (float*)d_out;

  char* ws = (char*)d_ws;
  size_t off = 0;
  float*  xnF    = (float*)(ws + off);  off = align256(off + (size_t)ROWS*D_*4);
  __bf16* xnB    = (__bf16*)(ws + off); off = align256(off + (size_t)ROWS*D_*2);
  __bf16* WvT    = (__bf16*)(ws + off); off = align256(off + (size_t)D_*H_*2);
  __bf16* WmxT   = (__bf16*)(ws + off); off = align256(off + (size_t)D_*MXW*2);
  __bf16* WhT    = (__bf16*)(ws + off); off = align256(off + (size_t)H_*D_*2);
  __bf16* vT     = (__bf16*)(ws + off); off = align256(off + (size_t)B_*H_*L_*2);
  __bf16* mx_inB = (__bf16*)(ws + off); off = align256(off + (size_t)ROWS*D_*2);
  __bf16* rwB    = (__bf16*)(ws + off); off = align256(off + (size_t)ROWS*D_*2);
  __bf16* qkB    = (__bf16*)(ws + off); off = align256(off + (size_t)ROWS*Z_*2);
  __bf16* gateB  = (__bf16*)(ws + off); off = align256(off + (size_t)ROWS*H_*2);
  __bf16* interB = (__bf16*)(ws + off); off = align256(off + (size_t)ROWS*D_*2);
  __bf16* qBuf   = (__bf16*)(ws + off); off = align256(off + (size_t)ROWS*Z_*2);
  __bf16* kBuf   = (__bf16*)(ws + off); off = align256(off + (size_t)ROWS*Z_*2);
  float*  biasF  = (float*)(ws + off);  off = align256(off + (size_t)C_*C_*4);
  __bf16* wgB    = (__bf16*)(ws + off); off = align256(off + (size_t)ROWS*H_*2);
  (void)ws_size; (void)n_in; (void)in_sizes; (void)out_size;

  (void)hipFuncSetAttribute((const void*)attn_kernel,
                            hipFuncAttributeMaxDynamicSharedMemorySize, ATTN_LDS_BYTES);

  // 1) weight conversions f32 -> bf16, transposed to [N][K]
  {
    size_t n1 = (size_t)D_*H_, n2 = (size_t)D_*MXW, n3 = (size_t)H_*D_;
    f2bf_t_kernel<<<(unsigned)((n1+255)/256), 256, 0, stream>>>(Wv,  WvT,  D_, H_);
    f2bf_t_kernel<<<(unsigned)((n2+255)/256), 256, 0, stream>>>(Wmx, WmxT, D_, MXW);
    f2bf_t_kernel<<<(unsigned)((n3+255)/256), 256, 0, stream>>>(Wh,  WhT,  H_, D_);
  }
  // 2) scalenorm
  scalenorm_kernel<<<ROWS, 256, 0, stream>>>(x, g, xnF, xnB);
  // 3) value = silu(xn @ Wv + bv)  -> vT[b][h][t]
  value_gemm_kernel<<<dim3(ROWS/BM, H_/BN), 256, 0, stream>>>(xnB, WvT, bv, vT);
  // 4) EMA recurrence + fused mx_in = silu(ema + xn*res_w)
  ema_scan_kernel<<<(B_*D_)/8, 256, 0, stream>>>(xnF, damping, decay, expans, kproj, res_w, mx_inB);
  // 5) base = mx_in @ Wmx + bmx with split epilogue
  mx_gemm_kernel<<<dim3(ROWS/BM, MXW/BN), 256, 0, stream>>>(mx_inB, WmxT, bmx,
                                                            rwB, qkB, gateB, interB);
  // 6) q/k affine
  qk_prep_kernel<<<(unsigned)(((size_t)ROWS*Z_)/256), 256, 0, stream>>>(qkB, qk_w, qk_b, qBuf, kBuf);
  // 7) rotary bias table
  bias_kernel<<<(C_*C_)/256, 256, 0, stream>>>(alpha, beta, biasF);
  // 8) chunked causal attention, fused *gate
  attn_kernel<<<dim3(B_*NCH, 4), 256, ATTN_LDS_BYTES, stream>>>(qBuf, kBuf, vT, gateB,
                                                                biasF, wgB);
  // 9) frame GEMM + fused silu / residual blend -> out
  frame_gemm_kernel<<<dim3(ROWS/BM, D_/BN), 256, 0, stream>>>(wgB, WhT, bh, interB, rwB, x, out);
}
